// UpLayer_81844896793192
// MI455X (gfx1250) — matrix operand order repaired and checked by reading, not compiled, not run on hardware
//
#include <hip/hip_runtime.h>
#include <math.h>

// ---------------------------------------------------------------------------
// Sizes (match reference)
// ---------------------------------------------------------------------------
#define N_P   65536
#define N_C   8192
#define NE    131072
#define D_P   80
#define D_C   48
#define D_ADD 16
#define D_EDGE 8
#define D_HID 64
#define F_IN  144          // D_P + D_C + D_ADD
#define KP1   160          // F_IN padded to multiple of 32 for WMMA K
#define NOUT  512          // D_EDGE * D_HID
#define D_OUT 144          // D_P + D_HID
#define TILE_E 32          // edges per block (two 16-row M-tiles per wave)
#define MT    2            // M-tiles per wave

typedef __bf16 v8bf  __attribute__((ext_vector_type(8)));
typedef __bf16 v16bf __attribute__((ext_vector_type(16)));
typedef float  v8f   __attribute__((ext_vector_type(8)));

// ---------------------------------------------------------------------------
// Kernel 1: out[n, 0:80] = x_p[n], out[n, 80:144] = 0   (atomic target init)
// ---------------------------------------------------------------------------
__global__ void gnn_init_out(const float* __restrict__ x_p,
                             float* __restrict__ out) {
  int idx = blockIdx.x * blockDim.x + threadIdx.x;
  if (idx >= N_P * D_OUT) return;
  int n = idx / D_OUT;
  int f = idx - n * D_OUT;
  out[idx] = (f < D_P) ? x_p[n * D_P + f] : 0.0f;
}

// ---------------------------------------------------------------------------
// Kernel 2: convert weights to bf16, transposed to [n][k] so each lane's
// B-fragment (16 consecutive K for fixed N) is one contiguous 32B load.
// Wt1: [512][160] (K zero-padded 144->160),  Wt2: [512][64].
// W1 flat layout: W1[f*512 + g*64 + h]  (f = K, n = g*64 + h)
// ---------------------------------------------------------------------------
__global__ void gnn_convert_w(const float* __restrict__ W1,
                              const float* __restrict__ W2,
                              __bf16* __restrict__ Wt1,
                              __bf16* __restrict__ Wt2) {
  int idx = blockIdx.x * blockDim.x + threadIdx.x;
  const int total1 = NOUT * KP1;        // 81920
  const int total2 = NOUT * D_HID;      // 32768
  if (idx < total1) {
    int n = idx / KP1;
    int k = idx - n * KP1;
    Wt1[idx] = (k < F_IN) ? (__bf16)W1[k * NOUT + n] : (__bf16)0.0f;
  } else if (idx < total1 + total2) {
    int i2 = idx - total1;
    int n = i2 / D_HID;
    int k = i2 - n * D_HID;
    Wt2[i2] = (__bf16)W2[k * NOUT + n];
  }
}

// ---------------------------------------------------------------------------
// Kernel 3: per-edge message pipeline on WMMA.
// 8 waves / block, TILE_E=32 edges per block, MT=2 M-tiles per wave.
// Wave w owns output columns [w*64, w*64+64) == g = w of the [E,8,64] tensor,
// so the edge_attr contraction is a per-wave scale + 8-way LDS reduction.
// ---------------------------------------------------------------------------
__device__ __forceinline__ v16bf make_afrag(const __bf16* base_lo,
                                            const __bf16* base_hi) {
  v8bf lo = *(const v8bf*)base_lo;
  v8bf hi = *(const v8bf*)base_hi;
  v16bf a;
#pragma unroll
  for (int i = 0; i < 8; ++i) { a[i] = lo[i]; a[i + 8] = hi[i]; }
  return a;
}

__launch_bounds__(256, 1)
__global__ void gnn_edge_msg(const float* __restrict__ xp,
                             const float* __restrict__ xc,
                             const int*   __restrict__ ei,     // [2][NE]
                             const float* __restrict__ eattr,  // [NE][8]
                             const float* __restrict__ eadd,   // [NE][16]
                             const float* __restrict__ b1,
                             const float* __restrict__ b2,
                             const __bf16* __restrict__ Wt1,   // [512][160]
                             const __bf16* __restrict__ Wt2,   // [512][64]
                             float* __restrict__ out) {
  __shared__ __bf16 sA[TILE_E * KP1];            // gathered bf16 input tile
  __shared__ __bf16 sM[TILE_E * D_HID];          // layer-1 activations (bf16)
  __shared__ float  sAttr[TILE_E * D_EDGE];
  __shared__ float  sStage[8 * TILE_E * D_HID];  // per-wave partials (64KB)
  __shared__ int    sSrc[TILE_E];
  __shared__ int    sDst[TILE_E];

  const int tid  = threadIdx.x;
  const int w    = tid >> 5;          // wave id 0..7  == g index
  const int lane = tid & 31;
  const int half = lane >> 4;         // 0: lanes 0-15, 1: lanes 16-31
  const int ln   = lane & 15;
  const int e0   = blockIdx.x * TILE_E;

  if (tid < TILE_E) {
    sSrc[tid] = ei[e0 + tid];         // indexes x_c
    sDst[tid] = ei[NE + e0 + tid];    // indexes x_p
  }
  sAttr[tid] = eattr[e0 * D_EDGE + tid];  // 32*8 == 256 == blockDim
  // warm L2/WGP$ for this wave's weight slab
  __builtin_prefetch(&Wt1[(w * 64 + ln) * KP1], 0, 1);
  __syncthreads();

  // ---- gather + f32->bf16 the [32 x 160] A tile (K padded with zeros) ----
#pragma unroll 4
  for (int idx = tid; idx < TILE_E * KP1; idx += 256) {
    int e = idx / KP1;
    int f = idx - e * KP1;
    float v = 0.0f;
    if (f < D_P)             v = xp[sDst[e] * D_P + f];
    else if (f < D_P + D_C)  v = xc[sSrc[e] * D_C + (f - D_P)];
    else if (f < F_IN)       v = eadd[(e0 + e) * D_ADD + (f - D_P - D_C)];
    sA[idx] = (__bf16)v;
  }
  __syncthreads();

  // attr value a[e, w] for the rows each accumulator element covers
  float av[MT][8];
#pragma unroll
  for (int mt = 0; mt < MT; ++mt)
#pragma unroll
    for (int r = 0; r < 8; ++r)
      av[mt][r] = sAttr[(mt * 16 + r + 8 * half) * D_EDGE + w];

  // =====================  GEMM1: [32x160] x [160x512]  =====================
  v8f acc[MT][4];
#pragma unroll
  for (int mt = 0; mt < MT; ++mt)
#pragma unroll
    for (int j = 0; j < 4; ++j)
#pragma unroll
      for (int i = 0; i < 8; ++i) acc[mt][j][i] = 0.0f;

#pragma unroll
  for (int ks = 0; ks < 5; ++ks) {
    const int k0 = ks * 32;
    // B-fragments once per k-step, reused across MT M-tiles
    v16bf b[4];
#pragma unroll
    for (int j = 0; j < 4; ++j) {
      const int n = w * 64 + j * 16 + ln;
      b[j] = *(const v16bf*)&Wt1[n * KP1 + k0 + half * 16];
    }
#pragma unroll
    for (int mt = 0; mt < MT; ++mt) {
      const int row = mt * 16 + ln;
      v16bf a = make_afrag(&sA[row * KP1 + k0 + half * 16],
                           &sA[row * KP1 + k0 + half * 16 + 8]);
#pragma unroll
      for (int j = 0; j < 4; ++j)
        acc[mt][j] = __builtin_amdgcn_wmma_f32_16x16x32_bf16(
            false, a, false, b[j], (short)0, acc[mt][j], false, false);
    }
  }

  // scale by a[e, w] and stage for cross-g reduction
#pragma unroll
  for (int mt = 0; mt < MT; ++mt)
#pragma unroll
    for (int j = 0; j < 4; ++j)
#pragma unroll
      for (int r = 0; r < 8; ++r) {
        int e = mt * 16 + r + 8 * half;
        int h = j * 16 + ln;
        sStage[(w * TILE_E + e) * D_HID + h] = av[mt][r] * acc[mt][j][r];
      }
  __syncthreads();

  // reduce over g, +b1, SiLU, convert to bf16 layer-2 A tile
#pragma unroll
  for (int idx = tid; idx < TILE_E * D_HID; idx += 256) {
    float s = 0.0f;
#pragma unroll
    for (int g = 0; g < 8; ++g) s += sStage[g * TILE_E * D_HID + idx];
    s += b1[idx & 63];
    s = s / (1.0f + __expf(-s));      // silu
    sM[idx] = (__bf16)s;
  }
  __syncthreads();

  // =====================  GEMM2: [32x64] x [64x512]  =======================
#pragma unroll
  for (int mt = 0; mt < MT; ++mt)
#pragma unroll
    for (int j = 0; j < 4; ++j)
#pragma unroll
      for (int i = 0; i < 8; ++i) acc[mt][j][i] = 0.0f;

#pragma unroll
  for (int ks = 0; ks < 2; ++ks) {
    const int k0 = ks * 32;
    v16bf b[4];
#pragma unroll
    for (int j = 0; j < 4; ++j) {
      const int n = w * 64 + j * 16 + ln;
      b[j] = *(const v16bf*)&Wt2[n * D_HID + k0 + half * 16];
    }
#pragma unroll
    for (int mt = 0; mt < MT; ++mt) {
      const int row = mt * 16 + ln;
      v16bf a = make_afrag(&sM[row * D_HID + k0 + half * 16],
                           &sM[row * D_HID + k0 + half * 16 + 8]);
#pragma unroll
      for (int j = 0; j < 4; ++j)
        acc[mt][j] = __builtin_amdgcn_wmma_f32_16x16x32_bf16(
            false, a, false, b[j], (short)0, acc[mt][j], false, false);
    }
  }

#pragma unroll
  for (int mt = 0; mt < MT; ++mt)
#pragma unroll
    for (int j = 0; j < 4; ++j)
#pragma unroll
      for (int r = 0; r < 8; ++r) {
        int e = mt * 16 + r + 8 * half;
        int h = j * 16 + ln;
        sStage[(w * TILE_E + e) * D_HID + h] = av[mt][r] * acc[mt][j][r];
      }
  __syncthreads();

  // reduce over g, +b2, SiLU, scatter-add into out[dst, 80+h]
#pragma unroll
  for (int idx = tid; idx < TILE_E * D_HID; idx += 256) {
    float s = 0.0f;
#pragma unroll
    for (int g = 0; g < 8; ++g) s += sStage[g * TILE_E * D_HID + idx];
    int e = idx >> 6;
    int h = idx & 63;
    s += b2[h];
    s = s / (1.0f + __expf(-s));
    atomicAdd(&out[sDst[e] * D_OUT + D_P + h], s);
  }
}

// ---------------------------------------------------------------------------
// Launcher
// ---------------------------------------------------------------------------
extern "C" void kernel_launch(void* const* d_in, const int* in_sizes, int n_in,
                              void* d_out, int out_size, void* d_ws, size_t ws_size,
                              hipStream_t stream) {
  const float* x_p   = (const float*)d_in[0];
  const float* x_c   = (const float*)d_in[1];
  const int*   ei    = (const int*)  d_in[2];
  const float* eattr = (const float*)d_in[3];
  // d_in[4] = batch (unused, norm=None)
  const float* eadd  = (const float*)d_in[5];
  const float* W1    = (const float*)d_in[6];
  const float* b1    = (const float*)d_in[7];
  const float* W2    = (const float*)d_in[8];
  const float* b2    = (const float*)d_in[9];
  float* out = (float*)d_out;

  // workspace: bf16 transposed weights
  __bf16* Wt1 = (__bf16*)d_ws;                       // 512*160*2 = 163840 B
  __bf16* Wt2 = (__bf16*)((char*)d_ws + NOUT * KP1 * sizeof(__bf16));

  // 1) weight convert/transpose (tiny)
  {
    int total = NOUT * KP1 + NOUT * D_HID;
    gnn_convert_w<<<(total + 255) / 256, 256, 0, stream>>>(W1, W2, Wt1, Wt2);
  }
  // 2) init output (residual copy + zero agg region)
  {
    int total = N_P * D_OUT;
    gnn_init_out<<<(total + 255) / 256, 256, 0, stream>>>(x_p, out);
  }
  // 3) edge message kernel: 32 edges per 256-thread (8-wave) block
  gnn_edge_msg<<<NE / TILE_E, 256, 0, stream>>>(x_p, x_c, ei, eattr, eadd,
                                                b1, b2, Wt1, Wt2, out);
}
